// static_heto_graph2_56581899157996
// MI455X (gfx1250) — compile-verified, matrix operand order checked by reading, not run on hardware
//
#include <hip/hip_runtime.h>
#include <hip/hip_bf16.h>
#include <math.h>

typedef __attribute__((ext_vector_type(16))) _Float16 v16h;
typedef __attribute__((ext_vector_type(8)))  _Float16 v8h;
typedef __attribute__((ext_vector_type(4)))  _Float16 v4h;
typedef __attribute__((ext_vector_type(8)))  float    v8f;

#define NW_N 100000
#define NT_N 800
#define ND_N 8000
#define B_N  16
#define H_N  128
#define HINP_N 300
#define KPAD_A 320   // 300 padded to 10 K-slices

// LDS byte offset of a __shared__-derived pointer (ISA 10.2: LDS addr = addr[31:0])
__device__ __forceinline__ unsigned lds_off(const void* p) {
  return (unsigned)(unsigned long long)p;
}
// async copy 16B global -> LDS (per active lane), tracked on ASYNCcnt
__device__ __forceinline__ void async_b128(unsigned loff, const void* gptr) {
  unsigned long long ga = (unsigned long long)gptr;
  asm volatile("global_load_async_to_lds_b128 %0, %1, off" :: "v"(loff), "v"(ga) : "memory");
}
__device__ __forceinline__ void wait_async0() {
  asm volatile("s_wait_asynccnt 0x0" ::: "memory");
}

// =====================================================================
// pack weights once: f32 [K][128] -> f16 transposed [128][Kpad]
//   WtA : W_adapt  128 x 320 (zero padded past K=300)
//   WtL : 10 mats (W1[0..4], W2[0..4]) each 128 x 128
// =====================================================================
__global__ __launch_bounds__(256) void pack_weights(
    const float* __restrict__ Wa, const float* __restrict__ W1,
    const float* __restrict__ W2, _Float16* __restrict__ WtA,
    _Float16* __restrict__ WtL)
{
  int i = blockIdx.x * 256 + threadIdx.x;
  if (i < H_N * KPAD_A) {
    int n = i / KPAD_A, k = i % KPAD_A;
    WtA[i] = (k < HINP_N) ? (_Float16)Wa[(size_t)k * H_N + n] : (_Float16)0.0f;
  }
  int j = i - H_N * KPAD_A;
  if (j >= 0 && j < 10 * H_N * H_N) {
    int mat = j / (H_N * H_N), r = j % (H_N * H_N);
    int n = r / H_N, k = r % H_N;
    const float* src = (mat < 5) ? (W1 + (size_t)mat * H_N * H_N)
                                 : (W2 + (size_t)(mat - 5) * H_N * H_N);
    WtL[j] = (_Float16)src[(size_t)k * H_N + n];
  }
}

// ---- shared GEMM epilogue: acc (f32) + bias -> LDS f16 tile -> coalesced store
__device__ __forceinline__ void epilogue_store(
    v8f acc[8], const float* __restrict__ bias, _Float16* Cs /*[128][136]*/,
    int tid, int lane, int wave, int row0, int M, _Float16* __restrict__ C)
{
  __syncthreads();                    // LDS (As/Ws) no longer needed
  const int hi = lane >> 4;
  #pragma unroll
  for (int nt = 0; nt < 8; ++nt) {
    const int n = nt * 16 + (lane & 15);
    const float bv = bias[n];
    #pragma unroll
    for (int r = 0; r < 8; ++r)
      Cs[(wave * 16 + hi * 8 + r) * 136 + n] = (_Float16)(acc[nt][r] + bv);
  }
  __syncthreads();
  const int mh = tid >> 1, cb = (tid & 1) * 64;
  if (row0 + mh < M) {
    _Float16* crow = C + (size_t)(row0 + mh) * H_N + cb;
    #pragma unroll
    for (int j = 0; j < 8; ++j)
      *(v8h*)(crow + j * 8) = *(const v8h*)&Cs[mh * 136 + cb + j * 8];
  }
}

// =====================================================================
// f16 GEMM, K=128: C[M,128] = A[M,128] @ W + bias   (A,W,C all f16)
// async-to-LDS staging; 8 waves x 8 col-tiles x 16x16x32 WMMA
// =====================================================================
__global__ __launch_bounds__(256) void gemm_h_k128(
    const _Float16* __restrict__ A, int M,
    const _Float16* __restrict__ Wt,    // [128][128] n-major (pre-transposed)
    const float* __restrict__ bias,
    _Float16* __restrict__ C)
{
  __shared__ __align__(16) unsigned char smem[34816];
  _Float16* As = (_Float16*)smem;              // [128][40]
  _Float16* Ws = (_Float16*)(smem + 10240);    // [128][40]
  _Float16* Cs = (_Float16*)smem;              // [128][136] (reused)

  const int tid = threadIdx.x, lane = tid & 31, wave = tid >> 5;
  const int row0 = blockIdx.x * 128;
  const int mh = tid >> 1, k0 = (tid & 1) * 16;
  const int rowA = (row0 + mh < M) ? (row0 + mh) : (M - 1);   // clamp, no divergence
  const _Float16* aRow = A + (size_t)rowA * H_N;
  const _Float16* wRow = Wt + (size_t)mh * H_N;

  const unsigned la = lds_off(As + mh * 40 + k0);
  const unsigned lw = lds_off(Ws + mh * 40 + k0);

  v8f acc[8] = {};

  for (int kc = 0; kc < 4; ++kc) {
    const int kb = kc << 5;
    __syncthreads();
    async_b128(la,      aRow + kb + k0);
    async_b128(la + 16, aRow + kb + k0 + 8);
    async_b128(lw,      wRow + kb + k0);
    async_b128(lw + 16, wRow + kb + k0 + 8);
    wait_async0();
    __syncthreads();

    const int mloc = wave * 16 + (lane & 15);
    const int hi   = lane >> 4;
    v8h a0 = *(const v8h*)&As[mloc * 40 + hi * 8];
    v8h a1 = *(const v8h*)&As[mloc * 40 + 16 + hi * 8];
    v16h afrag;
    #pragma unroll
    for (int i = 0; i < 8; ++i) { afrag[i] = a0[i]; afrag[8 + i] = a1[i]; }

    #pragma unroll
    for (int nt = 0; nt < 8; ++nt) {
      const int nn = nt * 16 + (lane & 15);
      v8h b0 = *(const v8h*)&Ws[nn * 40 + hi * 8];
      v8h b1 = *(const v8h*)&Ws[nn * 40 + 16 + hi * 8];
      v16h bfrag;
      #pragma unroll
      for (int i = 0; i < 8; ++i) { bfrag[i] = b0[i]; bfrag[8 + i] = b1[i]; }
      acc[nt] = __builtin_amdgcn_wmma_f32_16x16x32_f16(
          false, afrag, false, bfrag, (short)0, acc[nt], false, false);
    }
  }
  epilogue_store(acc, bias, Cs, tid, lane, wave, row0, M, C);
}

// =====================================================================
// adapt GEMM: C[M,128] = gather(A_f32[.,300])[word_ids] @ WtA + bias -> f16
// float4 staging + cvt; OOB guard only on the final K-slice
// =====================================================================
__global__ __launch_bounds__(256) void gemm_adapt(
    const float* __restrict__ A, const int* __restrict__ gidx, int M,
    const _Float16* __restrict__ WtA,   // [128][320] n-major, zero padded
    const float* __restrict__ bias,
    _Float16* __restrict__ C)
{
  __shared__ __align__(16) unsigned char smem[34816];
  _Float16* As = (_Float16*)smem;
  _Float16* Ws = (_Float16*)(smem + 10240);
  _Float16* Cs = (_Float16*)smem;

  const int tid = threadIdx.x, lane = tid & 31, wave = tid >> 5;
  const int row0 = blockIdx.x * 128;
  const int mh = tid >> 1, k0 = (tid & 1) * 16;
  const int rowA = (row0 + mh < M) ? (row0 + mh) : (M - 1);
  const float* aRow = A + (size_t)gidx[rowA] * HINP_N;
  const _Float16* wRow = WtA + (size_t)mh * KPAD_A;
  const unsigned lw = lds_off(Ws + mh * 40 + k0);

  v8f acc[8] = {};

  for (int kc = 0; kc < 10; ++kc) {
    const int kb = kc << 5;
    __syncthreads();
    // W slice: contiguous f16, padded -> unguarded async copy
    async_b128(lw,      wRow + kb + k0);
    async_b128(lw + 16, wRow + kb + k0 + 8);
    // A slice: f32 -> f16 convert through VGPRs
    v8h h0, h1;
    if (kc < 9) {                      // fully in bounds (k < 288+32 <= 320, data<300 only kc<9 ok: 288)
      const float4 f0 = *(const float4*)(aRow + kb + k0);
      const float4 f1 = *(const float4*)(aRow + kb + k0 + 4);
      const float4 f2 = *(const float4*)(aRow + kb + k0 + 8);
      const float4 f3 = *(const float4*)(aRow + kb + k0 + 12);
      h0[0]=(_Float16)f0.x; h0[1]=(_Float16)f0.y; h0[2]=(_Float16)f0.z; h0[3]=(_Float16)f0.w;
      h0[4]=(_Float16)f1.x; h0[5]=(_Float16)f1.y; h0[6]=(_Float16)f1.z; h0[7]=(_Float16)f1.w;
      h1[0]=(_Float16)f2.x; h1[1]=(_Float16)f2.y; h1[2]=(_Float16)f2.z; h1[3]=(_Float16)f2.w;
      h1[4]=(_Float16)f3.x; h1[5]=(_Float16)f3.y; h1[6]=(_Float16)f3.z; h1[7]=(_Float16)f3.w;
    } else {                           // last slice: k in [288,320), valid < 300
      #pragma unroll
      for (int i = 0; i < 8; ++i) {
        int ka = kb + k0 + i, kbx = kb + k0 + 8 + i;
        h0[i] = (ka  < HINP_N) ? (_Float16)aRow[ka]  : (_Float16)0.0f;
        h1[i] = (kbx < HINP_N) ? (_Float16)aRow[kbx] : (_Float16)0.0f;
      }
    }
    *(v8h*)&As[mh * 40 + k0]     = h0;
    *(v8h*)&As[mh * 40 + k0 + 8] = h1;
    wait_async0();
    __syncthreads();

    const int mloc = wave * 16 + (lane & 15);
    const int hi   = lane >> 4;
    v8h a0 = *(const v8h*)&As[mloc * 40 + hi * 8];
    v8h a1 = *(const v8h*)&As[mloc * 40 + 16 + hi * 8];
    v16h afrag;
    #pragma unroll
    for (int i = 0; i < 8; ++i) { afrag[i] = a0[i]; afrag[8 + i] = a1[i]; }

    #pragma unroll
    for (int nt = 0; nt < 8; ++nt) {
      const int nn = nt * 16 + (lane & 15);
      v8h b0 = *(const v8h*)&Ws[nn * 40 + hi * 8];
      v8h b1 = *(const v8h*)&Ws[nn * 40 + 16 + hi * 8];
      v16h bfrag;
      #pragma unroll
      for (int i = 0; i < 8; ++i) { bfrag[i] = b0[i]; bfrag[8 + i] = b1[i]; }
      acc[nt] = __builtin_amdgcn_wmma_f32_16x16x32_f16(
          false, afrag, false, bfrag, (short)0, acc[nt], false, false);
    }
  }
  epilogue_store(acc, bias, Cs, tid, lane, wave, row0, M, C);
}

// ------------------------- edge helpers ------------------------------
__global__ __launch_bounds__(256) void edge_count(
    const int* __restrict__ dst, int E, float* __restrict__ cnt)
{
  int e = blockIdx.x * 256 + threadIdx.x;
  if (e < E) unsafeAtomicAdd(&cnt[dst[e]], 1.0f);
}

__global__ __launch_bounds__(256) void invert_counts(float* __restrict__ c, int n)
{
  int i = blockIdx.x * 256 + threadIdx.x;
  if (i < n) c[i] = 1.0f / fmaxf(c[i], 1.0f);
}

// one wave per edge; f16 source, f32 atomic accumulation (4 floats/lane)
__global__ __launch_bounds__(256) void edge_scatter(
    const _Float16* __restrict__ h, const int* __restrict__ src,
    const int* __restrict__ dst, const float* __restrict__ w,
    int E, float* __restrict__ out)
{
  int e = blockIdx.x * 8 + (threadIdx.x >> 5);
  if (e >= E) return;
  int lane = threadIdx.x & 31;
  int u = src[e], v = dst[e];
  float we = w[e];
  v4h hv = *(const v4h*)(h + (size_t)u * H_N + lane * 4);
  float* o = out + (size_t)v * H_N + lane * 4;
  unsafeAtomicAdd(o + 0, (float)hv[0] * we);
  unsafeAtomicAdd(o + 1, (float)hv[1] * we);
  unsafeAtomicAdd(o + 2, (float)hv[2] * we);
  unsafeAtomicAdd(o + 3, (float)hv[3] * we);
}

// -------------------- gather / finalize / pool -----------------------
__global__ __launch_bounds__(256) void gather_rows_h(
    const float* __restrict__ emb, const int* __restrict__ ids,
    int n, _Float16* __restrict__ out)
{
  int i = blockIdx.x * 256 + threadIdx.x;
  if (i < n * H_N) out[i] = (_Float16)emb[(size_t)ids[i >> 7] * H_N + (i & 127)];
}

// out_f16 = relu(s * inv[row])
__global__ __launch_bounds__(256) void finalize1_h(
    const float* __restrict__ s, const float* __restrict__ inv,
    _Float16* __restrict__ out, int n)
{
  int i = blockIdx.x * 256 + threadIdx.x;
  if (i < n * H_N) out[i] = (_Float16)fmaxf(s[i] * inv[i >> 7], 0.0f);
}

// out_f16 = relu(s1*i1[row] + s2*i2[row])
__global__ __launch_bounds__(256) void finalize2_h(
    const float* __restrict__ s1, const float* __restrict__ i1,
    const float* __restrict__ s2, const float* __restrict__ i2,
    _Float16* __restrict__ out, int n)
{
  int i = blockIdx.x * 256 + threadIdx.x;
  if (i < n * H_N)
    out[i] = (_Float16)fmaxf(s1[i] * i1[i >> 7] + s2[i] * i2[i >> 7], 0.0f);
}

// f32 in-place: hd = relu(s1*i1 + s2*i2)
__global__ __launch_bounds__(256) void finalize2_f(
    float* __restrict__ s1, const float* __restrict__ i1,
    const float* __restrict__ s2, const float* __restrict__ i2, int n)
{
  int i = blockIdx.x * 256 + threadIdx.x;
  if (i < n * H_N)
    s1[i] = fmaxf(s1[i] * i1[i >> 7] + s2[i] * i2[i >> 7], 0.0f);
}

// graph g owns docs [g*500, (g+1)*500)
__global__ __launch_bounds__(128) void pool_max(
    const float* __restrict__ hd, float* __restrict__ pool)
{
  int g = blockIdx.x, n = threadIdx.x;
  const int per = ND_N / B_N;
  float m = -INFINITY;
  for (int d = g * per; d < (g + 1) * per; ++d)
    m = fmaxf(m, hd[(size_t)d * H_N + n]);
  pool[g * H_N + n] = m;
}

__global__ __launch_bounds__(512) void head_kernel(
    const float* __restrict__ pool, const float* __restrict__ Wout,
    const float* __restrict__ bout, const float* __restrict__ y,
    float* __restrict__ out /* [0]=loss, [1..16]=y_pred */)
{
  __shared__ float logits[B_N];
  int wave = threadIdx.x >> 5, lane = threadIdx.x & 31;
  float s = 0.0f;
  #pragma unroll
  for (int i = 0; i < 4; ++i) {
    int n = lane * 4 + i;
    s += pool[wave * H_N + n] * Wout[n];
  }
  #pragma unroll
  for (int off = 16; off; off >>= 1) s += __shfl_down(s, off, 32);
  if (lane == 0) logits[wave] = s + bout[0];
  __syncthreads();
  if (threadIdx.x < B_N) {
    float lg = logits[threadIdx.x];
    out[1 + threadIdx.x] = 1.0f / (1.0f + expf(-lg));
  }
  if (threadIdx.x == 0) {
    float loss = 0.0f;
    for (int g = 0; g < B_N; ++g) {
      float lg = logits[g];
      loss += fmaxf(lg, 0.0f) - lg * y[g] + log1pf(expf(-fabsf(lg)));
    }
    out[0] = loss / (float)B_N;
  }
}

// =====================================================================
extern "C" void kernel_launch(void* const* d_in, const int* in_sizes, int n_in,
                              void* d_out, int out_size, void* d_ws, size_t ws_size,
                              hipStream_t stream)
{
  const int*   word_ids  = (const int*)d_in[0];
  const int*   topic_ids = (const int*)d_in[1];
  const int*   ww_src = (const int*)d_in[3];  const int* ww_dst = (const int*)d_in[4];
  const float* ww_w   = (const float*)d_in[5];
  const int*   wt_src = (const int*)d_in[6];  const int* wt_dst = (const int*)d_in[7];
  const float* wt_w   = (const float*)d_in[8];
  const int*   wd_src = (const int*)d_in[9];  const int* wd_dst = (const int*)d_in[10];
  const float* wd_w   = (const float*)d_in[11];
  const int*   td_src = (const int*)d_in[12]; const int* td_dst = (const int*)d_in[13];
  const float* td_w   = (const float*)d_in[14];
  const int*   tt_src = (const int*)d_in[15]; const int* tt_dst = (const int*)d_in[16];
  const float* tt_w   = (const float*)d_in[17];
  const float* word_embeds  = (const float*)d_in[18];
  const float* topic_embeds = (const float*)d_in[19];
  const float* W_adapt = (const float*)d_in[20];
  const float* b_adapt = (const float*)d_in[21];
  const float* W1 = (const float*)d_in[22];
  const float* b1 = (const float*)d_in[23];
  const float* W2 = (const float*)d_in[24];
  const float* b2 = (const float*)d_in[25];
  const float* W_out = (const float*)d_in[26];
  const float* b_out = (const float*)d_in[27];
  const float* y     = (const float*)d_in[28];

  const int E_ww = in_sizes[3],  E_wt = in_sizes[6], E_wd = in_sizes[9];
  const int E_td = in_sizes[12], E_tt = in_sizes[15];

  // -------- workspace layout (256B aligned) --------
  char* ws = (char*)d_ws;
  size_t off = 0;
  auto alloc = [&](size_t bytes) -> void* {
    void* p = ws + off;
    off = (off + bytes + 255) & ~(size_t)255;
    return p;
  };
  _Float16* xw0 = (_Float16*)alloc((size_t)NW_N * H_N * 2);  // f16 activ ping
  _Float16* xw1 = (_Float16*)alloc((size_t)NW_N * H_N * 2);  // f16 activ pong
  _Float16* xt0 = (_Float16*)alloc((size_t)NT_N * H_N * 2);
  _Float16* xt1 = (_Float16*)alloc((size_t)NT_N * H_N * 2);
  float* s_ww = (float*)alloc((size_t)NW_N * H_N * 4);
  float* s_wt = (float*)alloc((size_t)NT_N * H_N * 4);
  float* s_tt = (float*)alloc((size_t)NT_N * H_N * 4);
  float* s_wd = (float*)alloc((size_t)ND_N * H_N * 4);       // becomes hd in place
  float* s_td = (float*)alloc((size_t)ND_N * H_N * 4);
  const int CNT_TOTAL = NW_N + NT_N + NT_N + ND_N + ND_N;
  float* cnt  = (float*)alloc((size_t)CNT_TOTAL * 4);
  float* inv_ww = cnt;
  float* inv_wt = inv_ww + NW_N;
  float* inv_tt = inv_wt + NT_N;
  float* inv_wd = inv_tt + NT_N;
  float* inv_td = inv_wd + ND_N;
  float* pool = (float*)alloc((size_t)B_N * H_N * 4);
  _Float16* WtA = (_Float16*)alloc((size_t)H_N * KPAD_A * 2);
  _Float16* WtL = (_Float16*)alloc((size_t)10 * H_N * H_N * 2);

  auto cdiv = [](int a, int b) { return (a + b - 1) / b; };
  const size_t MATB = (size_t)H_N * H_N;   // f16 elems per packed matrix

  // -------- weight packing + per-etype in-degree reciprocals --------
  pack_weights<<<cdiv(H_N * KPAD_A + 10 * H_N * H_N, 256), 256, 0, stream>>>(
      W_adapt, W1, W2, WtA, WtL);
  hipMemsetAsync(cnt, 0, (size_t)CNT_TOTAL * 4, stream);
  edge_count<<<cdiv(E_ww, 256), 256, 0, stream>>>(ww_dst, E_ww, inv_ww);
  edge_count<<<cdiv(E_wt, 256), 256, 0, stream>>>(wt_dst, E_wt, inv_wt);
  edge_count<<<cdiv(E_tt, 256), 256, 0, stream>>>(tt_dst, E_tt, inv_tt);
  edge_count<<<cdiv(E_wd, 256), 256, 0, stream>>>(wd_dst, E_wd, inv_wd);
  edge_count<<<cdiv(E_td, 256), 256, 0, stream>>>(td_dst, E_td, inv_td);
  invert_counts<<<cdiv(CNT_TOTAL, 256), 256, 0, stream>>>(cnt, CNT_TOTAL);

  // -------- input embeddings --------
  gather_rows_h<<<cdiv(NT_N * H_N, 256), 256, 0, stream>>>(topic_embeds, topic_ids, NT_N, xt0);
  gemm_adapt<<<cdiv(NW_N, 128), 256, 0, stream>>>(word_embeds, word_ids, NW_N, WtA, b_adapt, xw0);

  // ================= layer 1 (hd output dead -> skip wd/td) ==========
  gemm_h_k128<<<cdiv(NW_N, 128), 256, 0, stream>>>(xw0, NW_N, WtL + 0 * MATB, b1 + 0 * H_N, xw1);
  gemm_h_k128<<<cdiv(NW_N, 128), 256, 0, stream>>>(xw1, NW_N, WtL + 1 * MATB, b1 + 1 * H_N, xw0);
  gemm_h_k128<<<cdiv(NW_N, 128), 256, 0, stream>>>(xw0, NW_N, WtL + 2 * MATB, b1 + 2 * H_N, xw1); // final word h
  gemm_h_k128<<<cdiv(NT_N, 128), 256, 0, stream>>>(xt0, NT_N, WtL + 3 * MATB, b1 + 3 * H_N, xt1);
  gemm_h_k128<<<cdiv(NT_N, 128), 256, 0, stream>>>(xt1, NT_N, WtL + 4 * MATB, b1 + 4 * H_N, xt0); // final topic h

  hipMemsetAsync(s_ww, 0, (size_t)NW_N * H_N * 4, stream);
  hipMemsetAsync(s_wt, 0, (size_t)NT_N * H_N * 4, stream);
  hipMemsetAsync(s_tt, 0, (size_t)NT_N * H_N * 4, stream);
  edge_scatter<<<cdiv(E_ww, 8), 256, 0, stream>>>(xw1, ww_src, ww_dst, ww_w, E_ww, s_ww);
  edge_scatter<<<cdiv(E_wt, 8), 256, 0, stream>>>(xw1, wt_src, wt_dst, wt_w, E_wt, s_wt);
  edge_scatter<<<cdiv(E_tt, 8), 256, 0, stream>>>(xt0, tt_src, tt_dst, tt_w, E_tt, s_tt);
  finalize1_h<<<cdiv(NW_N * H_N, 256), 256, 0, stream>>>(s_ww, inv_ww, xw0, NW_N);          // hw L2-in
  finalize2_h<<<cdiv(NT_N * H_N, 256), 256, 0, stream>>>(s_wt, inv_wt, s_tt, inv_tt, xt1, NT_N);

  // ================= layer 2 (hw/ht outputs dead -> only wd/td) ======
  gemm_h_k128<<<cdiv(NW_N, 128), 256, 0, stream>>>(xw0, NW_N, WtL + 5 * MATB, b2 + 0 * H_N, xw1);
  gemm_h_k128<<<cdiv(NW_N, 128), 256, 0, stream>>>(xw1, NW_N, WtL + 6 * MATB, b2 + 1 * H_N, xw0);
  gemm_h_k128<<<cdiv(NW_N, 128), 256, 0, stream>>>(xw0, NW_N, WtL + 7 * MATB, b2 + 2 * H_N, xw1); // final word h
  gemm_h_k128<<<cdiv(NT_N, 128), 256, 0, stream>>>(xt1, NT_N, WtL + 8 * MATB, b2 + 3 * H_N, xt0);
  gemm_h_k128<<<cdiv(NT_N, 128), 256, 0, stream>>>(xt0, NT_N, WtL + 9 * MATB, b2 + 4 * H_N, xt1); // final topic h

  hipMemsetAsync(s_wd, 0, (size_t)ND_N * H_N * 4, stream);
  hipMemsetAsync(s_td, 0, (size_t)ND_N * H_N * 4, stream);
  edge_scatter<<<cdiv(E_wd, 8), 256, 0, stream>>>(xw1, wd_src, wd_dst, wd_w, E_wd, s_wd);
  edge_scatter<<<cdiv(E_td, 8), 256, 0, stream>>>(xt1, td_src, td_dst, td_w, E_td, s_td);
  finalize2_f<<<cdiv(ND_N * H_N, 256), 256, 0, stream>>>(s_wd, inv_wd, s_td, inv_td, ND_N); // hd

  // ================= readout =================
  pool_max<<<B_N, 128, 0, stream>>>(s_wd, pool);
  head_kernel<<<1, 512, 0, stream>>>(pool, W_out, b_out, y, (float*)d_out);
}
